// TLSMaskBuilder_26250840113229
// MI455X (gfx1250) — matrix operand order repaired and checked by reading, not compiled, not run on hardware
//
#include <hip/hip_runtime.h>
#include <hip/hip_bf16.h>

#define DEV_INLINE __device__ __forceinline__

typedef __bf16 bf16_t;
typedef __attribute__((ext_vector_type(16))) __bf16 v16bf;
typedef __attribute__((ext_vector_type(8)))  __bf16 v8bf;
typedef __attribute__((ext_vector_type(8)))  float  v8f;

constexpr int BB  = 2;
constexpr int S   = 2048;
constexpr int D   = 1024;
constexpr int H   = 16;
constexpr int L   = 675;        // max(1, int(0.33 * 2048))
constexpr int WIN = 128;
constexpr float ALPHA = 0.7f;
constexpr float EPSV  = 1e-8f;

// ---------------------------------------------------------------------------
// 1) float -> bf16 copy + per-row sum of squares
// ---------------------------------------------------------------------------
__global__ void cast_sq_kernel(const float* __restrict__ h,
                               bf16_t* __restrict__ hbf,
                               float* __restrict__ sq) {
  int row = blockIdx.x;            // b*S + s
  int t   = threadIdx.x;           // 256 threads, 4 elems each
  const float* hr = h + (size_t)row * D;
  bf16_t* br = hbf + (size_t)row * D;
  float acc = 0.f;
#pragma unroll
  for (int c = 0; c < 4; ++c) {
    int d = t * 4 + c;
    float v = hr[d];
    acc += v * v;
    br[d] = (bf16_t)v;
  }
  __shared__ float red[256];
  red[t] = acc;
  __syncthreads();
  for (int st = 128; st > 0; st >>= 1) {
    if (t < st) red[t] += red[t + st];
    __syncthreads();
  }
  if (t == 0) sq[row] = red[0];
}

// ---------------------------------------------------------------------------
// 2) centroid[b, d] = mean over s of h[b, s, d]
// ---------------------------------------------------------------------------
__global__ void centroid_kernel(const float* __restrict__ h,
                                float* __restrict__ centroid) {
  int g = blockIdx.x * blockDim.x + threadIdx.x;
  if (g >= BB * D) return;
  int b = g / D, d = g % D;
  const float* p = h + (size_t)b * S * D + d;
  float acc = 0.f;
  for (int s = 0; s < S; ++s) acc += p[(size_t)s * D];
  centroid[g] = acc * (1.0f / (float)S);
}

// ---------------------------------------------------------------------------
// 3) dist-to-centroid + per-head dist-to-head-centroid
//    (head centroid is just a 64-slice of the full centroid)
// ---------------------------------------------------------------------------
__global__ void distfeat_kernel(const float* __restrict__ h,
                                const float* __restrict__ centroid,
                                float* __restrict__ dc,    // [B,S]
                                float* __restrict__ hd) {  // [B,H,S]
  int row = blockIdx.x;            // b*S + s
  int b = row / S;
  int s = row % S;
  int t = threadIdx.x;             // 256
  const float* hr = h + (size_t)row * D;
  const float* cr = centroid + (size_t)b * D;
  float acc = 0.f;
#pragma unroll
  for (int c = 0; c < 4; ++c) {
    int d = t * 4 + c;
    float v = hr[d] - cr[d];
    acc += v * v;
  }
  __shared__ float red[256];
  red[t] = acc;
  __syncthreads();
  // threads t<16: head t covers d in [64t, 64t+64) == red[16t .. 16t+16)
  float headsum = 0.f;
  if (t < H) {
#pragma unroll
    for (int q = 0; q < 16; ++q) headsum += red[t * 16 + q];
  }
  __syncthreads();
  for (int st = 128; st > 0; st >>= 1) {
    if (t < st) red[t] += red[t + st];
    __syncthreads();
  }
  if (t < H) hd[((size_t)b * H + t) * S + s] = sqrtf(headsum);
  if (t == 0) dc[row] = sqrtf(red[0]);
}

// ---------------------------------------------------------------------------
// 4) generic per-row max
// ---------------------------------------------------------------------------
__global__ void rowmax_kernel(const float* __restrict__ in,
                              float* __restrict__ out, int rowlen) {
  int row = blockIdx.x;
  int t = threadIdx.x;
  const float* p = in + (size_t)row * rowlen;
  float m = -__builtin_inff();
  for (int i = t; i < rowlen; i += blockDim.x) m = fmaxf(m, p[i]);
  __shared__ float red[256];
  red[t] = m;
  __syncthreads();
  for (int st = 128; st > 0; st >>= 1) {
    if (t < st) red[t] = fmaxf(red[t], red[t + st]);
    __syncthreads();
  }
  if (t == 0) out[row] = red[0];
}

// ---------------------------------------------------------------------------
// 5) WMMA Gram -> squared-distance matrix.
//    One wave computes a 16x16 tile of d2[b] with K-loop over D in steps of 32
//    using v_wmma_f32_16x16x32_bf16. Fragment layouts (contiguous K, same order in A and B):
//      A 16x32 bf16: lane(0-15)=row M, K = (lane>>4)*16 + {0..15}
//      B 32x16 bf16: lane(0-15)=col N, K = (lane>>4)*16 + {0..15}
//    Both read contiguous-K from row-major h, so no transpose load needed.
// ---------------------------------------------------------------------------
__global__ void __launch_bounds__(32)
gram_d2_kernel(const bf16_t* __restrict__ hbf,
               const float* __restrict__ sq,
               float* __restrict__ d2) {
  int jt = blockIdx.x, it = blockIdx.y, b = blockIdx.z;
  int lane = threadIdx.x;
  int half = lane >> 4;
  int l15  = lane & 15;
  const bf16_t* base = hbf + (size_t)b * S * D;
  const bf16_t* rowA = base + (size_t)(it * 16 + l15) * D + half * 16;
  const bf16_t* rowB = base + (size_t)(jt * 16 + l15) * D + half * 16;

  v8f c = {};
  for (int k0 = 0; k0 < D; k0 += 32) {
    v8bf alo = *(const v8bf*)(rowA + k0);        // K = kb .. kb+7
    v8bf ahi = *(const v8bf*)(rowA + k0 + 8);    // K = kb+8 .. kb+15
    v16bf a;
#pragma unroll
    for (int q = 0; q < 8; ++q) { a[q] = alo[q]; a[8 + q] = ahi[q]; }
    v16bf bm = *(const v16bf*)(rowB + k0);       // 16 contiguous K
    if (k0 + 32 < D)
      __builtin_prefetch((const void*)(rowB + k0 + 32), 0, 3);
    c = __builtin_amdgcn_wmma_f32_16x16x32_bf16(
        false, a, false, bm, (short)0, c, false, false);
  }

  int jcol = jt * 16 + l15;
  float sqj = sq[b * S + jcol];
#pragma unroll
  for (int r = 0; r < 8; ++r) {
    int irow = it * 16 + half * 8 + r;           // C layout: M = r + 8*(lane>>4)
    float sqi = sq[b * S + irow];
    float v = sqi + sqj - 2.0f * c[r];
    d2[((size_t)b * S + irow) * S + jcol] = fmaxf(v, 0.0f);
  }
}

// ---------------------------------------------------------------------------
// 6) 6th-smallest d2 per row (k=5 nearest neighbor incl. self) -> diversity
// ---------------------------------------------------------------------------
DEV_INLINE void insert6(float* a, float v) {
  if (v >= a[5]) return;
  a[5] = v;
#pragma unroll
  for (int i = 5; i > 0; --i) {
    if (a[i] < a[i - 1]) { float tmp = a[i]; a[i] = a[i - 1]; a[i - 1] = tmp; }
  }
}

__global__ void knn6_kernel(const float* __restrict__ d2,
                            float* __restrict__ divv) {
  int row = blockIdx.x;            // b*S + i
  int t = threadIdx.x;             // 256
  const float* p = d2 + (size_t)row * S;
  float best[6];
#pragma unroll
  for (int q = 0; q < 6; ++q) best[q] = __builtin_inff();
  for (int j = t; j < S; j += 256) insert6(best, p[j]);
  __shared__ float lists[256 * 6];
#pragma unroll
  for (int q = 0; q < 6; ++q) lists[t * 6 + q] = best[q];
  __syncthreads();
  for (int st = 128; st > 0; st >>= 1) {
    if (t < st) {
      float* A  = &lists[t * 6];
      float* Bp = &lists[(t + st) * 6];
      float merged[6];
#pragma unroll
      for (int q = 0; q < 6; ++q) merged[q] = A[q];
#pragma unroll
      for (int q = 0; q < 6; ++q) insert6(merged, Bp[q]);
#pragma unroll
      for (int q = 0; q < 6; ++q) A[q] = merged[q];
    }
    __syncthreads();
  }
  if (t == 0) divv[row] = sqrtf(lists[5]);
}

// ---------------------------------------------------------------------------
// 7) combined score = 0.7*(0.7*pers + 0.3*div) + 0.3*head_dist_norm
// ---------------------------------------------------------------------------
__global__ void combined_kernel(const float* __restrict__ dc,
                                const float* __restrict__ divv,
                                const float* __restrict__ hd,
                                const float* __restrict__ maxP,
                                const float* __restrict__ maxDiv,
                                const float* __restrict__ maxHD,
                                float* __restrict__ comb) {
  int g = blockIdx.x * blockDim.x + threadIdx.x;   // B*H*S
  if (g >= BB * H * S) return;
  int s = g % S;
  int bh = g / S;
  int b = bh / H;
  float pers = dc[b * S + s] / (maxP[b] + EPSV);
  float dv   = divv[b * S + s] / (maxDiv[b] + EPSV);
  float tls  = ALPHA * pers + (1.0f - ALPHA) * dv;
  float hdn  = hd[g] / (maxHD[bh] + EPSV);
  comb[g] = 0.7f * tls + 0.3f * hdn;
}

// ---------------------------------------------------------------------------
// 8) exact top-L per (b,h): full bitonic sort of 2048 (val,idx) pairs in LDS.
//    Order: value desc, index asc on ties (matches jax.lax.top_k).
// ---------------------------------------------------------------------------
__global__ void __launch_bounds__(1024)
topk_kernel(const float* __restrict__ comb,
            unsigned char* __restrict__ is_lm,     // [B*H, S]
            float* __restrict__ lm_out) {          // [B*H, L] as float
  int bh = blockIdx.x;
  int t = threadIdx.x;                              // 1024
  __shared__ float vals[S];
  __shared__ int   idxs[S];
  const float* row = comb + (size_t)bh * S;
  for (int i = t; i < S; i += 1024) { vals[i] = row[i]; idxs[i] = i; }
  __syncthreads();
  for (int k = 2; k <= S; k <<= 1) {
    for (int j = k >> 1; j > 0; j >>= 1) {
      for (int base = 0; base < S; base += 1024) {
        int i = base + t;
        int ixj = i ^ j;
        if (ixj > i) {
          float va = vals[i], vb = vals[ixj];
          int ia = idxs[i], ib = idxs[ixj];
          bool a_better = (va > vb) || (va == vb && ia < ib);
          bool up = ((i & k) == 0);
          bool doswap = up ? !a_better : a_better;
          if (doswap) {
            vals[i] = vb; vals[ixj] = va;
            idxs[i] = ib; idxs[ixj] = ia;
          }
        }
      }
      __syncthreads();
    }
  }
  unsigned char* flags = is_lm + (size_t)bh * S;
  for (int i = t; i < S; i += 1024) flags[i] = 0;
  __syncthreads();
  for (int i = t; i < L; i += 1024) {
    int p = idxs[i];
    flags[p] = 1;
    lm_out[(size_t)bh * L + i] = (float)p;
  }
}

// ---------------------------------------------------------------------------
// 9) mask[b,h,i,j] = (local_window OR lm_row OR lm_col) AND causal
//    Vectorized: each thread computes 4 consecutive j and issues one
//    16-byte store (global_store_b128) -> 537 MB written at full HBM width.
//    Column flags for the 4 j's come from a single aligned u32 load.
// ---------------------------------------------------------------------------
__global__ void mask_kernel(const unsigned char* __restrict__ is_lm,
                            float* __restrict__ out) {
  size_t g = (size_t)blockIdx.x * blockDim.x + threadIdx.x;  // group of 4 j
  int j0 = (int)(g & (size_t)(S / 4 - 1)) * 4;
  int i  = (int)((g >> 9) & (size_t)(S - 1));
  int bh = (int)(g >> 20);
  const unsigned char* f = is_lm + (size_t)bh * S;
  bool fi = f[i] != 0;
  unsigned int fj4 = *(const unsigned int*)(f + j0);          // 4 column flags
  float4 r;
  float* rp = &r.x;
#pragma unroll
  for (int c = 0; c < 4; ++c) {
    int j = j0 + c;
    int diff = i - j;
    bool local = (diff >= 0) && (diff < WIN);
    bool fj = ((fj4 >> (8 * c)) & 0xFFu) != 0;
    bool v = (local || fi || fj) && (i >= j);
    rp[c] = v ? 1.0f : 0.0f;
  }
  *(float4*)(out + g * 4) = r;
}

// ---------------------------------------------------------------------------
extern "C" void kernel_launch(void* const* d_in, const int* in_sizes, int n_in,
                              void* d_out, int out_size, void* d_ws, size_t ws_size,
                              hipStream_t stream) {
  (void)in_sizes; (void)n_in; (void)out_size; (void)ws_size;
  const float* h = (const float*)d_in[0];
  float* out = (float*)d_out;

  // workspace carve-out (256B aligned slabs), total ~43 MB
  char* p = (char*)d_ws;
  auto alloc = [&](size_t bytes) -> char* {
    char* r = p;
    p += (bytes + 255) & ~(size_t)255;
    return r;
  };
  bf16_t* hbf      = (bf16_t*)alloc((size_t)BB * S * D * sizeof(bf16_t));
  float*  sq       = (float*) alloc((size_t)BB * S * sizeof(float));
  float*  centroid = (float*) alloc((size_t)BB * D * sizeof(float));
  float*  dc       = (float*) alloc((size_t)BB * S * sizeof(float));
  float*  hd       = (float*) alloc((size_t)BB * H * S * sizeof(float));
  float*  d2       = (float*) alloc((size_t)BB * S * S * sizeof(float));
  float*  divv     = (float*) alloc((size_t)BB * S * sizeof(float));
  float*  maxP     = (float*) alloc((size_t)BB * sizeof(float));
  float*  maxDiv   = (float*) alloc((size_t)BB * sizeof(float));
  float*  maxHD    = (float*) alloc((size_t)BB * H * sizeof(float));
  float*  comb     = (float*) alloc((size_t)BB * H * S * sizeof(float));
  unsigned char* is_lm = (unsigned char*)alloc((size_t)BB * H * S);

  const size_t maskElems = (size_t)BB * H * S * S;

  cast_sq_kernel  <<<BB * S, 256, 0, stream>>>(h, hbf, sq);
  centroid_kernel <<<(BB * D + 255) / 256, 256, 0, stream>>>(h, centroid);
  distfeat_kernel <<<BB * S, 256, 0, stream>>>(h, centroid, dc, hd);

  gram_d2_kernel  <<<dim3(S / 16, S / 16, BB), 32, 0, stream>>>(hbf, sq, d2);

  knn6_kernel     <<<BB * S, 256, 0, stream>>>(d2, divv);
  rowmax_kernel   <<<BB, 256, 0, stream>>>(dc, maxP, S);
  rowmax_kernel   <<<BB, 256, 0, stream>>>(divv, maxDiv, S);
  rowmax_kernel   <<<BB * H, 256, 0, stream>>>(hd, maxHD, S);

  combined_kernel <<<(BB * H * S + 255) / 256, 256, 0, stream>>>(
      dc, divv, hd, maxP, maxDiv, maxHD, comb);

  topk_kernel     <<<BB * H, 1024, 0, stream>>>(comb, is_lm, out + maskElems);

  // 4 elements per thread -> maskElems/4 threads
  mask_kernel     <<<(unsigned)(maskElems / 4 / 256), 256, 0, stream>>>(is_lm, out);
}
